// CS_Attn_28200755265611
// MI455X (gfx1250) — matrix operand order repaired and checked by reading, not compiled, not run on hardware
//
#include <hip/hip_runtime.h>
#include <hip/hip_bf16.h>

// ---------------------------------------------------------------------------
// Problem constants (from reference)
// ---------------------------------------------------------------------------
#define BB   2
#define SS   128
#define SEG  128
#define DD   768
#define RLn  16
#define HH   8
#define DKk  64
#define DVv  96
#define NRPF 96
#define NN   2048            // S*RL attention tokens
#define RTOT 4095            // 2N-1 rel rows
#define RPAD 4224            // padded rel rows (33*128), rows >= 4095 are zero

typedef _Float16 half16 __attribute__((ext_vector_type(16)));
typedef _Float16 half8  __attribute__((ext_vector_type(8)));
typedef float    float8 __attribute__((ext_vector_type(8)));

__device__ __forceinline__ int laneid() { return (int)(threadIdx.x & 31); }

__device__ __forceinline__ float8 zero8() {
  float8 z;
#pragma unroll
  for (int t = 0; t < 8; ++t) z[t] = 0.f;
  return z;
}

// WMMA wrapper: D = A(16x32 f16) * B(32x16 f16) + C(16x16 f32)
__device__ __forceinline__ float8 wmma32(half16 a, half16 b, float8 c) {
  return __builtin_amdgcn_wmma_f32_16x16x32_f16(
      /*neg_a=*/false, a, /*neg_b=*/false, b,
      /*c_mod=*/(short)0, c, /*reuse_a=*/false, /*reuse_b=*/false);
}

// ---------------------------------------------------------------------------
// CDNA5 async copy to LDS (GLOBAL_LOAD_ASYNC_TO_LDS_B128, tracked by ASYNCcnt)
// Builtin signature (from probe diagnostics): (v4i AS1* src, v4i AS3* dst,
// const int offset, const int cpol)
// ---------------------------------------------------------------------------
typedef int v4i_t __attribute__((vector_size(16)));
typedef __attribute__((address_space(1))) v4i_t gv4i;
typedef __attribute__((address_space(3))) v4i_t lv4i;

__device__ __forceinline__ void async_ld16(const _Float16* g, _Float16* l) {
#if __has_builtin(__builtin_amdgcn_global_load_async_to_lds_b128)
  __builtin_amdgcn_global_load_async_to_lds_b128((gv4i*)g, (lv4i*)l,
                                                 /*offset=*/0, /*cpol=*/0);
#else
  *(half8*)l = *(const half8*)g;   // synchronous fallback (still LDS staging)
#endif
}

__device__ __forceinline__ void wait_async() {
#if __has_builtin(__builtin_amdgcn_s_wait_asynccnt)
  __builtin_amdgcn_s_wait_asynccnt(0);
#else
  __builtin_amdgcn_s_waitcnt(0);   // conservative fallback
#endif
}

// A-fragment (16x32, f16 row-major source, row stride ld elements).
// Lane layout per ISA: lanes 0-15 -> rows 0-15 with K {0..7,16..23},
// lanes 16-31 -> rows 0-15 with K {8..15,24..31}.
__device__ __forceinline__ half16 afrag16(const _Float16* A, int ld, int kb) {
  int l = laneid();
  const _Float16* p = A + (size_t)(l & 15) * ld + kb + ((l < 16) ? 0 : 8);
  half8 lo = *(const half8*)p;
  half8 hi = *(const half8*)(p + 16);
  half16 r;
#pragma unroll
  for (int t = 0; t < 8; ++t) { r[t] = lo[t]; r[8 + t] = hi[t]; }
  return r;
}

// A-fragment from an fp32 source with row bounds guard (rows >= nrow -> 0).
__device__ __forceinline__ half16 afrag_f32(const float* A, int ld, int row0,
                                            int kb, int nrow) {
  int l = laneid();
  int row = row0 + (l & 15);
  int ks = (l < 16) ? 0 : 8;
  half16 r;
#pragma unroll
  for (int t = 0; t < 16; ++t) r[t] = (_Float16)0.f;
  if (row < nrow) {
    const float* p = A + (size_t)row * ld + kb + ks;
#pragma unroll
    for (int t = 0; t < 8; ++t) {
      r[t]     = (_Float16)p[t];
      r[8 + t] = (_Float16)p[16 + t];
    }
  }
  return r;
}

// B-fragment (32x16) from a column-major store Bcm[col][k] (k contiguous,
// col stride ld). Lanes 0-15 -> cols 0-15 with K {0..7,16..23}, lanes 16-31 K {8..15,24..31}.
__device__ __forceinline__ half16 bfrag16(const _Float16* Bcm, int ld, int kb) {
  int l = laneid();
  const _Float16* p = Bcm + (size_t)(l & 15) * ld + kb + ((l < 16) ? 0 : 8);
  half8 lo = *(const half8*)p;
  half8 hi = *(const half8*)(p + 16);
  half16 r;
#pragma unroll
  for (int t = 0; t < 8; ++t) { r[t] = lo[t]; r[8 + t] = hi[t]; }
  return r;
}

// ---------------------------------------------------------------------------
// K1: slice reps + layernorm -> h16 (B*N x 768, f16)
// ---------------------------------------------------------------------------
__global__ void k_ln(const float* __restrict__ x, const float* __restrict__ g,
                     const float* __restrict__ bta, _Float16* __restrict__ h16) {
  int row = blockIdx.x;                   // b*N + n
  int b = row >> 11, n = row & 2047;
  int s = n >> 4, rl = n & 15;
  const float* xp = x + ((((size_t)b * SS + s) * SEG) + rl) * DD;

  int w = (int)(threadIdx.x >> 5), l = laneid();
  float vals[3];
  float sum = 0.f, sq = 0.f;
#pragma unroll
  for (int i = 0; i < 3; ++i) {
    float v = xp[threadIdx.x + 256 * i];
    vals[i] = v; sum += v; sq += v * v;
  }
#pragma unroll
  for (int m = 16; m >= 1; m >>= 1) {
    sum += __shfl_xor(sum, m, 32);
    sq  += __shfl_xor(sq, m, 32);
  }
  __shared__ float rsum[8], rsq[8];
  if (l == 0) { rsum[w] = sum; rsq[w] = sq; }
  __syncthreads();
  float tot = 0.f, totq = 0.f;
#pragma unroll
  for (int wv = 0; wv < 8; ++wv) { tot += rsum[wv]; totq += rsq[wv]; }
  float mu = tot * (1.f / 768.f);
  float var = totq * (1.f / 768.f) - mu * mu;
  float rstd = rsqrtf(var + 1e-5f);
#pragma unroll
  for (int i = 0; i < 3; ++i) {
    int d = (int)threadIdx.x + 256 * i;
    h16[(size_t)row * DD + d] = (_Float16)((vals[i] - mu) * rstd * g[d] + bta[d]);
  }
}

// ---------------------------------------------------------------------------
// K2: pack weights to f16, transposed (col-major: W_T[col][k])
// ---------------------------------------------------------------------------
__global__ void k_pack(const float* __restrict__ Wq, const float* __restrict__ Wk,
                       const float* __restrict__ Wv, const float* __restrict__ Wr,
                       const float* __restrict__ Wo,
                       _Float16* __restrict__ qkvT, _Float16* __restrict__ wrT,
                       _Float16* __restrict__ woT) {
  long i = (long)blockIdx.x * 256 + threadIdx.x;
  const long n_qkv = 1792L * 768, n_wr = 512L * 96, n_wo = 768L * 768;
  if (i < n_qkv) {
    int col = (int)(i / 768), k = (int)(i % 768);
    float v;
    if (col < 512)       v = Wq[(size_t)k * 512 + col];
    else if (col < 1024) v = Wk[(size_t)k * 512 + (col - 512)];
    else                 v = Wv[(size_t)k * 768 + (col - 1024)];
    qkvT[i] = (_Float16)v;
  } else if (i < n_qkv + n_wr) {
    long j = i - n_qkv;
    int col = (int)(j / 96), k = (int)(j % 96);
    wrT[j] = (_Float16)Wr[(size_t)k * 512 + col];
  } else if (i < n_qkv + n_wr + n_wo) {
    long j = i - n_qkv - n_wr;
    int col = (int)(j / 768), k = (int)(j % 768);
    woT[j] = (_Float16)Wo[(size_t)k * 768 + col];
  }
}

// ---------------------------------------------------------------------------
// K3: rel_k = pos_emb(4095x96) @ Wr -> relk[h][r][dk] f16 (rows>=4095 zero)
// ---------------------------------------------------------------------------
__global__ void k_relk(const float* __restrict__ pos_emb,
                       const _Float16* __restrict__ wrT,
                       _Float16* __restrict__ relk) {
  int w = (int)(threadIdx.x >> 5), l = laneid();
  int r0 = blockIdx.x * 128 + w * 16;
  int c0 = blockIdx.y * 64;               // head = blockIdx.y
  float8 acc[4];
#pragma unroll
  for (int ct = 0; ct < 4; ++ct) acc[ct] = zero8();

  for (int ks = 0; ks < 96; ks += 32) {
    half16 a = afrag_f32(pos_emb, NRPF, r0, ks, RTOT);
#pragma unroll
    for (int ct = 0; ct < 4; ++ct) {
      half16 b = bfrag16(wrT + (size_t)(c0 + ct * 16) * 96, 96, ks);
      acc[ct] = wmma32(a, b, acc[ct]);
    }
  }
  int rofs = (l < 16) ? 0 : 8;
#pragma unroll
  for (int ct = 0; ct < 4; ++ct) {
#pragma unroll
    for (int r = 0; r < 8; ++r) {
      int row = r0 + rofs + r;
      int dk = ct * 16 + (l & 15);
      relk[((size_t)blockIdx.y * RPAD + row) * DKk + dk] = (_Float16)acc[ct][r];
    }
  }
}

// ---------------------------------------------------------------------------
// K4: fused QKV GEMM h16(4096x768) @ WqkvT(->1792 cols)
//     The 96KB weight panel (64 cols x 768 k, contiguous) is async-staged to
//     LDS once per block and reused by all 8 waves and all 24 K-steps.
// ---------------------------------------------------------------------------
__global__ void k_qkv(const _Float16* __restrict__ h16,
                      const _Float16* __restrict__ wT,
                      const float* __restrict__ rwb, const float* __restrict__ rrb,
                      _Float16* __restrict__ qw, _Float16* __restrict__ qr,
                      _Float16* __restrict__ kk, _Float16* __restrict__ vv) {
  extern __shared__ char smem[];           // 96KB weight panel
  _Float16* Ws = (_Float16*)smem;
  int w = (int)(threadIdx.x >> 5), l = laneid();
  int row0 = blockIdx.x * 128 + w * 16;
  int c0 = blockIdx.y * 64;

  // async stage: 64 cols * 768 k * 2B = 98304 B = 6144 x b128
  {
    const _Float16* gp = wT + (size_t)c0 * DD;
    for (int i = threadIdx.x; i < 6144; i += 256)
      async_ld16(gp + (size_t)i * 8, Ws + (size_t)i * 8);
    wait_async();
    __syncthreads();
  }

  float8 acc[4];
#pragma unroll
  for (int ct = 0; ct < 4; ++ct) acc[ct] = zero8();

  const _Float16* A = h16 + (size_t)row0 * DD;
  for (int ks = 0; ks < DD; ks += 32) {
    half16 a = afrag16(A, DD, ks);
#pragma unroll
    for (int ct = 0; ct < 4; ++ct) {
      half16 b = bfrag16(Ws + (size_t)(ct * 16) * DD, DD, ks);
      acc[ct] = wmma32(a, b, acc[ct]);
    }
  }
  int rofs = (l < 16) ? 0 : 8;
#pragma unroll
  for (int ct = 0; ct < 4; ++ct) {
    int c = c0 + ct * 16 + (l & 15);
#pragma unroll
    for (int r = 0; r < 8; ++r) {
      int row = row0 + rofs + r;
      int b = row >> 11, n = row & 2047;
      float v = acc[ct][r];
      if (c < 512) {
        int h = c >> 6, dk = c & 63;
        size_t o = (((size_t)(b * HH + h) * NN) + n) * DKk + dk;
        float qs = v * 0.125f;                   // SCALE = DK^-0.5
        qw[o] = (_Float16)(qs + rwb[h * DKk + dk]);
        qr[o] = (_Float16)(qs + rrb[h * DKk + dk]);
      } else if (c < 1024) {
        int c2 = c - 512; int h = c2 >> 6, dk = c2 & 63;
        kk[(((size_t)(b * HH + h) * NN) + n) * DKk + dk] = (_Float16)v;
      } else {
        int c2 = c - 1024; int h = c2 / DVv, dv = c2 % DVv;
        vv[(((size_t)(b * HH + h) * DVv) + dv) * NN + n] = (_Float16)v;
      }
    }
  }
}

// ---------------------------------------------------------------------------
// K5: fused flash attention with relative shift.
//     Double-buffered async staging of the shared K (128x64) and V (96x128)
//     tiles into LDS; all content/PV WMMA B-fragments read from LDS.
// ---------------------------------------------------------------------------
__device__ __forceinline__ void stage_kv(const _Float16* kp, const _Float16* vp,
                                         int kb0, _Float16* Kb, _Float16* Vb) {
  // K tile: contiguous 128 rows x 64 halfs = 16KB = 1024 x b128
  const _Float16* kg = kp + (size_t)kb0 * DKk;
  for (int i = threadIdx.x; i < 1024; i += 256)
    async_ld16(kg + (size_t)i * 8, Kb + (size_t)i * 8);
  // V tile: 96 rows (dv) x 128 halfs (keys), global row stride NN
  for (int c = threadIdx.x; c < 1536; c += 256) {
    int row = c >> 4, off = (c & 15) * 8;
    async_ld16(vp + (size_t)row * NN + kb0 + off, Vb + row * 128 + off);
  }
}

__global__ void k_attn(const _Float16* __restrict__ qw16,
                       const _Float16* __restrict__ qr16,
                       const _Float16* __restrict__ k16,
                       const _Float16* __restrict__ v16,
                       const _Float16* __restrict__ relk,
                       _Float16* __restrict__ o16) {
  extern __shared__ char smem[];
  // layout: Kbuf[2]=32768B | Vbuf[2]=49152B | E=73728B | P=32768B  (184KB)
  _Float16* Kbuf = (_Float16*)smem;
  _Float16* Vbuf = (_Float16*)(smem + 32768);
  int w = (int)(threadIdx.x >> 5), l = laneid();
  float*    Ew = (float*)(smem + 81920) + (size_t)w * 16 * 144;
  _Float16* Pw = (_Float16*)(smem + 155648) + (size_t)w * 16 * 128;

  int bh = blockIdx.x >> 4;               // 16 query-blocks per (b,h)
  int qb = blockIdx.x & 15;
  int b = bh >> 3, h = bh & 7;
  int qstart = qb * 128 + w * 16;

  const _Float16* qwp = qw16 + (size_t)bh * NN * DKk;
  const _Float16* qrp = qr16 + (size_t)bh * NN * DKk;
  const _Float16* kp  = k16  + (size_t)bh * NN * DKk;
  const _Float16* vp  = v16  + (size_t)bh * DVv * NN;
  const _Float16* rp  = relk + (size_t)h * RPAD * DKk;

  half16 aw0 = afrag16(qwp + (size_t)qstart * DKk, DKk, 0);
  half16 aw1 = afrag16(qwp + (size_t)qstart * DKk, DKk, 32);
  half16 ar0 = afrag16(qrp + (size_t)qstart * DKk, DKk, 0);
  half16 ar1 = afrag16(qrp + (size_t)qstart * DKk, DKk, 32);

  float m[8], lsum[8];
#pragma unroll
  for (int r = 0; r < 8; ++r) { m[r] = -1e30f; lsum[r] = 0.f; }
  float8 O[6];
#pragma unroll
  for (int ct = 0; ct < 6; ++ct) O[ct] = zero8();

  int rofs = (l < 16) ? 0 : 8;
  int coln = l & 15;

  // prime buffer 0
  stage_kv(kp, vp, 0, Kbuf, Vbuf);
  wait_async();
  __syncthreads();

#pragma unroll 1
  for (int kb0 = 0; kb0 < NN; kb0 += 128) {
    int cur = (kb0 >> 7) & 1;
    _Float16* Kb = Kbuf + cur * 8192;
    _Float16* Vb = Vbuf + cur * 12288;
    // prefetch next tile into the other buffer while we compute
    if (kb0 + 128 < NN) {
      stage_kv(kp, vp, kb0 + 128, Kbuf + (cur ^ 1) * 8192,
               Vbuf + (cur ^ 1) * 12288);
      __builtin_prefetch(rp + (size_t)(NN - 15 + kb0 + 128 - qstart) * DKk, 0, 1);
    }
    // ---- relative band: E = q_r(16x64) @ rel_k[base-15 .. base+128]^T ----
    int base = (NN - 1) + kb0 - qstart;
    int rb = base - 15;                   // >= 0 always
#pragma unroll
    for (int rt = 0; rt < 9; ++rt) {
      float8 e = zero8();
      const _Float16* bb = rp + (size_t)(rb + rt * 16) * DKk;
      e = wmma32(ar0, bfrag16(bb, DKk, 0), e);
      e = wmma32(ar1, bfrag16(bb, DKk, 32), e);
#pragma unroll
      for (int r = 0; r < 8; ++r)
        Ew[(size_t)(rofs + r) * 144 + rt * 16 + coln] = e[r];
    }
    // ---- content scores (K from LDS) + shifted rel gather ----
    float8 S[8];
#pragma unroll
    for (int jt = 0; jt < 8; ++jt) {
      float8 s = zero8();
      const _Float16* bb = Kb + (size_t)(jt * 16) * DKk;
      s = wmma32(aw0, bfrag16(bb, DKk, 0), s);
      s = wmma32(aw1, bfrag16(bb, DKk, 32), s);
#pragma unroll
      for (int r = 0; r < 8; ++r) {
        int ii = rofs + r;
        int jj = jt * 16 + coln;
        s[r] += Ew[(size_t)ii * 144 + (jj - ii + 15)];
      }
      S[jt] = s;
    }
    // ---- online softmax per row (row = reg + 8*half) ----
#pragma unroll
    for (int r = 0; r < 8; ++r) {
      float mx = -1e30f;
#pragma unroll
      for (int jt = 0; jt < 8; ++jt) mx = fmaxf(mx, S[jt][r]);
#pragma unroll
      for (int sh = 8; sh >= 1; sh >>= 1) mx = fmaxf(mx, __shfl_xor(mx, sh, 32));
      float mn = fmaxf(m[r], mx);
      float sc = __expf(m[r] - mn);
      lsum[r] *= sc;
#pragma unroll
      for (int ct = 0; ct < 6; ++ct) O[ct][r] *= sc;
      float ps = 0.f;
#pragma unroll
      for (int jt = 0; jt < 8; ++jt) {
        float p = __expf(S[jt][r] - mn);
        ps += p;
        Pw[(size_t)(rofs + r) * 128 + jt * 16 + coln] = (_Float16)p;
      }
#pragma unroll
      for (int sh = 8; sh >= 1; sh >>= 1) ps += __shfl_xor(ps, sh, 32);
      lsum[r] += ps;
      m[r] = mn;
    }
    // ---- O += P(16x128) @ V(128x96), V tile in LDS as [dv][key] ----
#pragma unroll
    for (int ks = 0; ks < 4; ++ks) {
      half16 a = afrag16(Pw, 128, ks * 32);
#pragma unroll
      for (int ct = 0; ct < 6; ++ct) {
        half16 bv = bfrag16(Vb + (size_t)(ct * 16) * 128, 128, ks * 32);
        O[ct] = wmma32(a, bv, O[ct]);
      }
    }
    // close the pipeline stage: our async loads done, everyone done reading
    wait_async();
    __syncthreads();
  }
  // ---- normalize + write (B, N, H*DV) f16 ----
#pragma unroll
  for (int ct = 0; ct < 6; ++ct) {
#pragma unroll
    for (int r = 0; r < 8; ++r) {
      int row = qstart + rofs + r;
      int dv = ct * 16 + coln;
      o16[((size_t)b * NN + row) * (HH * DVv) + h * DVv + dv] =
          (_Float16)(O[ct][r] / lsum[r]);
    }
  }
}

// ---------------------------------------------------------------------------
// K6: output projection o16(4096x768) @ Wo + bo + residual, scatter into out
//     Weight panel async-staged to LDS (same pattern as K4).
// ---------------------------------------------------------------------------
__global__ void k_out(const _Float16* __restrict__ o16,
                      const _Float16* __restrict__ woT,
                      const float* __restrict__ bo, const float* __restrict__ x,
                      float* __restrict__ out) {
  extern __shared__ char smem[];
  _Float16* Ws = (_Float16*)smem;
  int w = (int)(threadIdx.x >> 5), l = laneid();
  int row0 = blockIdx.x * 128 + w * 16;
  int c0 = blockIdx.y * 64;

  {
    const _Float16* gp = woT + (size_t)c0 * DD;
    for (int i = threadIdx.x; i < 6144; i += 256)
      async_ld16(gp + (size_t)i * 8, Ws + (size_t)i * 8);
    wait_async();
    __syncthreads();
  }

  float8 acc[4];
#pragma unroll
  for (int ct = 0; ct < 4; ++ct) acc[ct] = zero8();

  const _Float16* A = o16 + (size_t)row0 * DD;
  for (int ks = 0; ks < DD; ks += 32) {
    half16 a = afrag16(A, DD, ks);
#pragma unroll
    for (int ct = 0; ct < 4; ++ct) {
      half16 b = bfrag16(Ws + (size_t)(ct * 16) * DD, DD, ks);
      acc[ct] = wmma32(a, b, acc[ct]);
    }
  }
  int rofs = (l < 16) ? 0 : 8;
#pragma unroll
  for (int ct = 0; ct < 4; ++ct) {
    int c = c0 + ct * 16 + (l & 15);
#pragma unroll
    for (int r = 0; r < 8; ++r) {
      int row = row0 + rofs + r;
      int b = row >> 11, n = row & 2047;
      int s = n >> 4, rl = n & 15;
      size_t idx = ((((size_t)b * SS + s) * SEG) + rl) * DD + c;
      out[idx] = acc[ct][r] + bo[c] + x[idx];
    }
  }
}

// ---------------------------------------------------------------------------
// K7: pass-through for non-representative tokens (seg rows 16..127)
// ---------------------------------------------------------------------------
__global__ void k_tail(const float* __restrict__ x, float* __restrict__ out) {
  size_t i = (size_t)blockIdx.x * 256 + threadIdx.x;
  if (i >= (size_t)BB * SS * (SEG - RLn) * DD) return;
  int d = (int)(i % DD);
  size_t t = i / DD;
  int rr = (int)(t % (SEG - RLn)) + RLn;
  t /= (SEG - RLn);
  int s = (int)(t % SS);
  int b = (int)(t / SS);
  size_t idx = ((((size_t)b * SS + s) * SEG) + rr) * DD + d;
  out[idx] = x[idx];
}

// ---------------------------------------------------------------------------
// launch
// ---------------------------------------------------------------------------
extern "C" void kernel_launch(void* const* d_in, const int* in_sizes, int n_in,
                              void* d_out, int out_size, void* d_ws, size_t ws_size,
                              hipStream_t stream) {
  const float* x    = (const float*)d_in[0];
  const float* ln_g = (const float*)d_in[1];
  const float* ln_b = (const float*)d_in[2];
  const float* Wq   = (const float*)d_in[3];
  const float* Wk   = (const float*)d_in[4];
  const float* Wv   = (const float*)d_in[5];
  const float* Wr   = (const float*)d_in[6];
  const float* rwb  = (const float*)d_in[7];
  const float* rrb  = (const float*)d_in[8];
  const float* Wo   = (const float*)d_in[9];
  const float* bo   = (const float*)d_in[10];
  const float* pos  = (const float*)d_in[11];
  float* out = (float*)d_out;

  char* ws = (char*)d_ws;
  _Float16* h16   = (_Float16*)(ws + 0);          //  6,291,456 B
  _Float16* qkvT  = (_Float16*)(ws + 6291456);    //  2,752,512 B
  _Float16* wrT   = (_Float16*)(ws + 9043968);    //     98,304 B
  _Float16* woT   = (_Float16*)(ws + 9142272);    //  1,179,648 B
  _Float16* relk  = (_Float16*)(ws + 10321920);   //  4,325,376 B
  _Float16* qw    = (_Float16*)(ws + 14647296);   //  4,194,304 B
  _Float16* qr    = (_Float16*)(ws + 18841600);   //  4,194,304 B
  _Float16* kk    = (_Float16*)(ws + 23035904);   //  4,194,304 B
  _Float16* vv    = (_Float16*)(ws + 27230208);   //  6,291,456 B
  _Float16* o16   = (_Float16*)(ws + 33521664);   //  6,291,456 B -> 39,813,120 total

  k_ln<<<BB * NN, 256, 0, stream>>>(x, ln_g, ln_b, h16);
  k_pack<<<7872, 256, 0, stream>>>(Wq, Wk, Wv, Wr, Wo, qkvT, wrT, woT);
  k_relk<<<dim3(RPAD / 128, HH), 256, 0, stream>>>(pos, wrT, relk);
  k_qkv<<<dim3(BB * NN / 128, 1792 / 64), 256, 98304, stream>>>(h16, qkvT, rwb,
                                                                rrb, qw, qr, kk, vv);
  size_t attn_lds = 188416;  // 2*(16K+24K) staging + 72K E + 32K P
  k_attn<<<BB * HH * (NN / 128), 256, attn_lds, stream>>>(qw, qr, kk, vv, relk, o16);
  k_out<<<dim3(BB * NN / 128, DD / 64), 256, 98304, stream>>>(o16, woT, bo, x, out);
  k_tail<<<(BB * SS * (SEG - RLn) * DD + 255) / 256, 256, 0, stream>>>(x, out);
}
